// PairRankGNN_26044681683635
// MI455X (gfx1250) — matrix operand order repaired and checked by reading, not compiled, not run on hardware
//
#include <hip/hip_runtime.h>
#include <hip/hip_bf16.h>
#include <math.h>

// ---------------------------------------------------------------------------
// PairRankGNN for MI455X (gfx1250, wave32, WMMA).
//
// Roofline: dense GEMMs ~56 GFLOP (trivial vs WMMA ceiling); dominant cost is
// 4x edge gather/scatter-add (1.6M edges x 256 f32 ~ 13 GB, largely L2-
// resident since h = 102 MB < 192 MB L2). Strategy: f32 in memory, in-register
// f32->f16 conversion into v_wmma_f32_16x16x32_f16 with f32 accumulation.
// N is a template parameter so B-fragment loads become immediate-offset
// clauses instead of serialized 64-bit address-add chains; two independent
// accumulator chains hide WMMA D->C latency.
// ---------------------------------------------------------------------------

#define N_NODES    100000
#define N_EDGES    1600000
#define UNITS      256
#define N_LAYERS   4
#define NUM_GRAPHS 512
#define IN_FEAT    9

typedef __attribute__((ext_vector_type(16))) _Float16 v16h;
typedef __attribute__((ext_vector_type(8)))  float    v8f;

// One full 16x16x32 WMMA step (vector path; K-aligned).
// A 16-bit 16x32 layout: lane half 'hi' holds K = abase..abase+7 (elems 0..7)
// and K = abase+16..abase+23 (elems 8..15), abase = k0 + hi*8.
// B 32x16 layout: same K order as A (elems 0..7: abase+e, elems 8..15: abase+16+e-8).
template<int N>
__device__ __forceinline__ v8f wmma_step(const float* __restrict__ A, size_t arow,
                                         const float* __restrict__ B, int col,
                                         int k0, int hi, v8f c)
{
    v16h a, b;
    const int abase = k0 + hi * 8;
    const float4* ap = reinterpret_cast<const float4*>(A + arow + abase);
    const float4 f0 = ap[0];
    const float4 f1 = ap[1];
    const float4 f2 = ap[4];              // +16 floats
    const float4 f3 = ap[5];
    a[0]  = (_Float16)f0.x; a[1]  = (_Float16)f0.y;
    a[2]  = (_Float16)f0.z; a[3]  = (_Float16)f0.w;
    a[4]  = (_Float16)f1.x; a[5]  = (_Float16)f1.y;
    a[6]  = (_Float16)f1.z; a[7]  = (_Float16)f1.w;
    a[8]  = (_Float16)f2.x; a[9]  = (_Float16)f2.y;
    a[10] = (_Float16)f2.z; a[11] = (_Float16)f2.w;
    a[12] = (_Float16)f3.x; a[13] = (_Float16)f3.y;
    a[14] = (_Float16)f3.z; a[15] = (_Float16)f3.w;

    // Compile-time N => single base address + 16 immediate-offset loads.
    const float* bp = B + (size_t)(k0 + hi * 8) * N + col;
#pragma unroll
    for (int e = 0; e < 16; ++e) b[e] = (_Float16)bp[(size_t)(e < 8 ? e : e + 8) * N];

    return __builtin_amdgcn_wmma_f32_16x16x32_f16(
        false, a, false, b, (short)0, c, false, false);
}

// Guarded tail step (zero-pads K remainder; used for the K=9 first layer).
template<int N>
__device__ __forceinline__ v8f wmma_step_tail(const float* __restrict__ A, size_t arow,
                                              const float* __restrict__ B, int col,
                                              int k0, int K, int hi, v8f c)
{
    v16h a, b;
    const int abase = k0 + hi * 8;
#pragma unroll
    for (int i = 0; i < 8; ++i) {
        const int ka = abase + i;
        const int kc = abase + 16 + i;
        a[i]     = (ka < K) ? (_Float16)A[arow + ka] : (_Float16)0.0f;
        a[8 + i] = (kc < K) ? (_Float16)A[arow + kc] : (_Float16)0.0f;
    }
    const int kb = k0 + hi * 8;
#pragma unroll
    for (int e = 0; e < 16; ++e) {
        const int k = kb + (e < 8 ? e : e + 8);
        b[e] = (k < K) ? (_Float16)B[(size_t)k * N + col] : (_Float16)0.0f;
    }
    return __builtin_amdgcn_wmma_f32_16x16x32_f16(
        false, a, false, b, (short)0, c, false, false);
}

// ---------------------------------------------------------------------------
// WMMA GEMM: C[M,N] = act(A[M,K] @ B[K,N] + bias[N])
// One 16x16 C tile per wave; 8 waves / 256-thread block cover 128 columns.
// ---------------------------------------------------------------------------
template<int N, int ACT>
__global__ void wmma_gemm_kernel(const float* __restrict__ A,
                                 const float* __restrict__ B,
                                 const float* __restrict__ bias,
                                 float* __restrict__ C,
                                 int M, int K)
{
    const int lane  = threadIdx.x & 31;
    const int wave  = threadIdx.x >> 5;           // 0..7
    const int tileM = blockIdx.x;
    const int tileN = blockIdx.y * 8 + wave;
    if (tileN * 16 >= N) return;                  // whole-wave exit, EXEC stays all-1s

    const int l16 = lane & 15;
    const int hi  = lane >> 4;                    // lane half
    int row = tileM * 16 + l16;
    if (row >= M) row = M - 1;                    // clamp (loads only; stores guarded)
    const int col = tileN * 16 + l16;
    const size_t arow = (size_t)row * K;

    // C/D layout: VGPR v holds (M = v + hi*8, N = l16) -> one column per lane.
    v8f c0, c1;
    const float bv = bias ? bias[col] : 0.0f;
#pragma unroll
    for (int i = 0; i < 8; ++i) { c0[i] = bv; c1[i] = 0.0f; }

    int k0 = 0;
    // Two independent accumulation chains per iteration (ILP across WMMAs).
    for (; k0 + 64 <= K; k0 += 64) {
        c0 = wmma_step<N>(A, arow, B, col, k0,      hi, c0);
        c1 = wmma_step<N>(A, arow, B, col, k0 + 32, hi, c1);
    }
    for (; k0 + 32 <= K; k0 += 32)
        c0 = wmma_step<N>(A, arow, B, col, k0, hi, c0);
    if (k0 < K)
        c0 = wmma_step_tail<N>(A, arow, B, col, k0, K, hi, c0);

#pragma unroll
    for (int v = 0; v < 8; ++v) {
        const int r = tileM * 16 + v + hi * 8;
        if (r < M) {
            float val = c0[v] + c1[v];
            if (ACT) val = tanhf(val);
            C[(size_t)r * N + col] = val;
        }
    }
}

// ---------------------------------------------------------------------------
// Elementwise / graph kernels
// ---------------------------------------------------------------------------
__global__ void fill_kernel(float* __restrict__ p, float v, long long n)
{
    long long i = (long long)blockIdx.x * blockDim.x + threadIdx.x;
    if (i < n) p[i] = v;
}

__global__ void deg_count_kernel(const int* __restrict__ dst,
                                 float* __restrict__ deg, int E)
{
    int e = blockIdx.x * blockDim.x + threadIdx.x;
    if (e < E) atomicAdd(&deg[dst[e]], 1.0f);
}

__global__ void rsqrt_kernel(float* __restrict__ p, int n)
{
    int i = blockIdx.x * blockDim.x + threadIdx.x;
    if (i < n) p[i] = rsqrtf(p[i]);
}

__global__ void edge_norm_kernel(const int* __restrict__ src,
                                 const int* __restrict__ dst,
                                 const float* __restrict__ dinv,
                                 float* __restrict__ nrm, int E)
{
    int e = blockIdx.x * blockDim.x + threadIdx.x;
    if (e < E) nrm[e] = dinv[src[e]] * dinv[dst[e]];
}

// agg = h * dinv^2  (self-loop term; also fully initializes agg each layer)
__global__ void init_agg_kernel(const float* __restrict__ h,
                                const float* __restrict__ dinv,
                                float* __restrict__ agg, long long total)
{
    long long i = (long long)blockIdx.x * blockDim.x + threadIdx.x;
    if (i >= total) return;
    const float d = dinv[(int)(i >> 8)];   // 256 feats per node
    agg[i] = h[i] * d * d;
}

// agg[dst] += h[src] * norm  — 64 threads per edge, float4 gather + 4 atomics.
// Consecutive lanes hit consecutive float4s of the same source row: coalesced.
__global__ void edge_scatter_kernel(const float* __restrict__ h,
                                    const int* __restrict__ src,
                                    const int* __restrict__ dst,
                                    const float* __restrict__ nrm,
                                    float* __restrict__ agg, int E)
{
    long long idx = (long long)blockIdx.x * blockDim.x + threadIdx.x;
    const int e = (int)(idx >> 6);
    if (e >= E) return;
    const int f = ((int)idx & 63) << 2;    // feature base 0..252
    const int s = src[e];
    const int d = dst[e];
    const float w = nrm[e];
    const float4 v = *reinterpret_cast<const float4*>(h + (size_t)s * 256 + f);
    float* o = agg + (size_t)d * 256 + f;
    atomicAdd(o + 0, v.x * w);
    atomicAdd(o + 1, v.y * w);
    atomicAdd(o + 2, v.z * w);
    atomicAdd(o + 3, v.w * w);
}

// x = tanh(agg + bias)  (in place)
__global__ void bias_tanh_kernel(float* __restrict__ agg,
                                 const float* __restrict__ bias, long long total)
{
    long long i = (long long)blockIdx.x * blockDim.x + threadIdx.x;
    if (i < total) agg[i] = tanhf(agg[i] + bias[i & 255]);
}

// h3[i] = h2[i,:] . fc3_w + fc3_b   (K=32, N=1)
__global__ void fc3_kernel(const float* __restrict__ h2,
                           const float* __restrict__ w,
                           const float* __restrict__ b,
                           float* __restrict__ h3, int n)
{
    int i = blockIdx.x * blockDim.x + threadIdx.x;
    if (i >= n) return;
    float s = b[0];
#pragma unroll
    for (int j = 0; j < 32; ++j) s += h2[(size_t)i * 32 + j] * w[j];
    h3[i] = s;
}

__global__ void pool_accum_kernel(const float* __restrict__ h3,
                                  const int* __restrict__ batch,
                                  float* __restrict__ sums,
                                  float* __restrict__ cnts, int n)
{
    int i = blockIdx.x * blockDim.x + threadIdx.x;
    if (i >= n) return;
    const int g = batch[i];
    atomicAdd(&sums[g], h3[i]);
    atomicAdd(&cnts[g], 1.0f);
}

__global__ void pool_final_kernel(const float* __restrict__ sums,
                                  const float* __restrict__ cnts,
                                  float* __restrict__ out, int g)
{
    int i = blockIdx.x * blockDim.x + threadIdx.x;
    if (i >= g) return;
    const float c = fmaxf(cnts[i], 1.0f);
    const float v = sums[i] / c;
    out[i] = 1.0f / (1.0f + expf(-v));
}

// ---------------------------------------------------------------------------
// Host orchestration
// ---------------------------------------------------------------------------
static inline void launch_gemm(const float* A, const float* B, const float* bias,
                               float* C, int M, int K, int N, int act,
                               hipStream_t s)
{
    dim3 grid((M + 15) / 16, ((N / 16) + 7) / 8);
    if (N == 256 && act == 0)
        wmma_gemm_kernel<256, 0><<<grid, 256, 0, s>>>(A, B, bias, C, M, K);
    else if (N == 256)
        wmma_gemm_kernel<256, 1><<<grid, 256, 0, s>>>(A, B, bias, C, M, K);
    else
        wmma_gemm_kernel<32, 1><<<grid, 256, 0, s>>>(A, B, bias, C, M, K);
}

extern "C" void kernel_launch(void* const* d_in, const int* in_sizes, int n_in,
                              void* d_out, int out_size, void* d_ws, size_t ws_size,
                              hipStream_t stream)
{
    (void)in_sizes; (void)n_in; (void)out_size; (void)ws_size;

    const float* x     = (const float*)d_in[0];    // [N_NODES, 9]
    const int*   edge  = (const int*)d_in[1];      // [2, E]
    const int*   src   = edge;
    const int*   dst   = edge + N_EDGES;
    const int*   batch = (const int*)d_in[2];      // [N_NODES]
    const float* Win   = (const float*)d_in[3];    // [9, 256]
    const float* bin   = (const float*)d_in[4];    // [256]
    const float* Ws    = (const float*)d_in[5];    // [3, 256, 256]
    const float* bs    = (const float*)d_in[6];    // [3, 256]
    const float* fc1w  = (const float*)d_in[7];    // [256, 256]
    const float* fc1b  = (const float*)d_in[8];
    const float* fc2w  = (const float*)d_in[9];    // [256, 32]
    const float* fc2b  = (const float*)d_in[10];
    const float* fc3w  = (const float*)d_in[11];   // [32, 1]
    const float* fc3b  = (const float*)d_in[12];
    float* out = (float*)d_out;                    // [512]

    // ---- workspace carve-out -------------------------------------------------
    char* ws = (char*)d_ws;
    const size_t NB = (size_t)N_NODES * 256 * sizeof(float);   // 102.4 MB
    float* pA   = (float*)(ws);                 // node buffer A
    float* pH   = (float*)(ws + NB);            // node buffer H (GEMM out)
    float* pC   = (float*)(ws + 2 * NB);        // node buffer C
    float* dinv = (float*)(ws + 3 * NB);
    float* nrm  = (float*)(ws + 3 * NB + ((size_t)N_NODES * 4 + 255) / 256 * 256);
    float* sums = nrm + N_EDGES;
    float* cnts = sums + NUM_GRAPHS;

    const long long TOT = (long long)N_NODES * 256;

    // ---- degrees / edge norms (recomputed every call: deterministic) --------
    fill_kernel<<<(N_NODES + 255) / 256, 256, 0, stream>>>(dinv, 1.0f, N_NODES);
    deg_count_kernel<<<(N_EDGES + 255) / 256, 256, 0, stream>>>(dst, dinv, N_EDGES);
    rsqrt_kernel<<<(N_NODES + 255) / 256, 256, 0, stream>>>(dinv, N_NODES);
    edge_norm_kernel<<<(N_EDGES + 255) / 256, 256, 0, stream>>>(src, dst, dinv, nrm, N_EDGES);

    // ---- 4 GCN conv layers ---------------------------------------------------
    const long long scat_threads = (long long)N_EDGES * 64;
    const int scat_blocks = (int)((scat_threads + 255) / 256);

    const float* X = x;          // layer input (f32)
    for (int layer = 0; layer < N_LAYERS; ++layer) {
        const float* W; const float* b; int K;
        if (layer == 0) { W = Win; b = bin; K = IN_FEAT; }
        else            { W = Ws + (size_t)(layer - 1) * UNITS * UNITS;
                          b = bs + (size_t)(layer - 1) * UNITS; K = UNITS; }

        float* agg = (layer & 1) ? pA : pC;      // ping-pong, X alternates too

        // h = X @ W         (WMMA)
        launch_gemm(X, W, nullptr, pH, N_NODES, K, UNITS, /*act=*/0, stream);
        // agg = h * dinv^2  (self loop + full init)
        init_agg_kernel<<<(int)((TOT + 255) / 256), 256, 0, stream>>>(pH, dinv, agg, TOT);
        // agg[dst] += h[src] * norm
        edge_scatter_kernel<<<scat_blocks, 256, 0, stream>>>(pH, src, dst, nrm, agg, N_EDGES);
        // X_next = tanh(agg + b)
        bias_tanh_kernel<<<(int)((TOT + 255) / 256), 256, 0, stream>>>(agg, b, TOT);

        X = agg;
    }
    // After 4 layers X == pA; pH and pC are free.

    // ---- MLP head ------------------------------------------------------------
    launch_gemm(pA, fc1w, fc1b, pC, N_NODES, UNITS, UNITS, /*act=*/1, stream); // tanh
    launch_gemm(pC, fc2w, fc2b, pH, N_NODES, UNITS, 32,    /*act=*/1, stream); // tanh
    float* h3 = pA;                                                            // reuse
    fc3_kernel<<<(N_NODES + 255) / 256, 256, 0, stream>>>(pH, fc3w, fc3b, h3, N_NODES);

    // ---- global mean pool + sigmoid -----------------------------------------
    fill_kernel<<<(2 * NUM_GRAPHS + 255) / 256, 256, 0, stream>>>(sums, 0.0f, 2 * NUM_GRAPHS);
    pool_accum_kernel<<<(N_NODES + 255) / 256, 256, 0, stream>>>(h3, batch, sums, cnts, N_NODES);
    pool_final_kernel<<<(NUM_GRAPHS + 255) / 256, 256, 0, stream>>>(sums, cnts, out, NUM_GRAPHS);
}
